// SatGNN_53979148976671
// MI455X (gfx1250) — matrix operand order repaired and checked by reading, not compiled, not run on hardware
//
#include <hip/hip_runtime.h>
#include <hip/hip_bf16.h>
#include <math.h>

typedef __attribute__((ext_vector_type(16))) _Float16 v16h;
typedef __attribute__((ext_vector_type(8)))  float    v8f;

#define D 128          // feature dim (D_IN == HIDDEN == 128)
#define NEG_SLOPE 0.2f

// ---------------------------------------------------------------------------
// Pack a 128x128 row-major f32 weight matrix into the WMMA B-operand layout
// for v_wmma_f32_16x16x32_f16 (wave32):
//   index = ((nt*4 + kt)*32 + lane)*16 + t
//   element t of lane L holds W[kt*32 + 8*(L>>4) + (t&7) + 16*(t>>3)][nt*16 + (L&15)]
// ---------------------------------------------------------------------------
__global__ void pack_w_kernel(const float* __restrict__ W, _Float16* __restrict__ Wp) {
    int idx = blockIdx.x * blockDim.x + threadIdx.x;
    if (idx >= D * D) return;
    int t    = idx & 15;
    int lane = (idx >> 4) & 31;
    int kt   = (idx >> 9) & 3;
    int nt   = idx >> 11;
    int k = kt * 32 + 8 * (lane >> 4) + (t & 7) + 16 * (t >> 3);
    int n = nt * 16 + (lane & 15);
    Wp[idx] = (_Float16)W[k * D + n];
}

// v[k] = sum_n W[k][n] * att[n]   (folds the attention projection: a = x @ (W@att))
__global__ void watt_kernel(const float* __restrict__ W, const float* __restrict__ att,
                            float* __restrict__ v) {
    int k = blockIdx.x * blockDim.x + threadIdx.x;
    if (k >= D) return;
    float s = 0.f;
    #pragma unroll 4
    for (int n = 0; n < D; ++n) s += W[k * D + n] * att[n];
    v[k] = s;
}

// ---------------------------------------------------------------------------
// WMMA GEMM: Y[nrows x 128] = X[nrows x 128] @ W (W pre-packed f16, B layout).
// One wave computes a 16-row stripe across all 128 output columns:
// 8 N-tiles x 4 K-steps of v_wmma_f32_16x16x32_f16, A fragments reused.
// nrows must be a multiple of 16 (50000 and 64 both are).
// ---------------------------------------------------------------------------
__global__ __launch_bounds__(256) void gemm_wmma_kernel(
        const float* __restrict__ X, const _Float16* __restrict__ Wp,
        float* __restrict__ Y, int nrows) {
    const int lane  = threadIdx.x & 31;
    const int wave  = threadIdx.x >> 5;
    const int tile  = blockIdx.x * 8 + wave;
    const int tiles = nrows >> 4;
    if (tile >= tiles) return;           // whole wave exits together (EXEC stays all-1 for WMMA)

    const int m0    = tile * 16;
    const int khalf = lane >> 4;         // lane half selects K sub-chunk (A layout)
    const int rowA  = m0 + (lane & 15);
    const float* xr = X + (size_t)rowA * D;

    // A fragments for the 4 K-tiles: elements 0..7 -> K = kt*32 + 8*khalf + t,
    // elements 8..15 -> K = kt*32 + 16 + 8*khalf + t (matches 16-bit A layout).
    v16h afrag[4];
    #pragma unroll
    for (int kt = 0; kt < 4; ++kt) {
        const float* p0 = xr + kt * 32 + 8 * khalf;
        const float* p1 = p0 + 16;
        #pragma unroll
        for (int t = 0; t < 8; ++t) {
            afrag[kt][t]     = (_Float16)p0[t];
            afrag[kt][t + 8] = (_Float16)p1[t];
        }
    }

    const int nc = lane & 15;
    #pragma unroll
    for (int nt = 0; nt < 8; ++nt) {
        v8f acc = {};
        #pragma unroll
        for (int kt = 0; kt < 4; ++kt) {
            v16h b = *(const v16h*)(Wp + ((size_t)(nt * 4 + kt) * 32 + lane) * 16);
            acc = __builtin_amdgcn_wmma_f32_16x16x32_f16(
                      false, afrag[kt], false, b, (short)0, acc, false, false);
        }
        // D layout: VGPR r -> row m0 + 8*khalf + r, col nt*16 + (lane&15)
        float* yp = Y + (size_t)(m0 + 8 * khalf) * D + nt * 16 + nc;
        #pragma unroll
        for (int r = 0; r < 8; ++r) yp[(size_t)r * D] = acc[r];
    }
}

// a_src[i] = x_i . vs ; a_dst[i] = x_i . vd  (one wave per node, float4/lane)
__global__ __launch_bounds__(256) void node_scores_kernel(
        const float* __restrict__ X, const float* __restrict__ vs,
        const float* __restrict__ vd, float* __restrict__ as_,
        float* __restrict__ ad_, int n) {
    int node = (int)((blockIdx.x * (size_t)blockDim.x + threadIdx.x) >> 5);
    int lane = threadIdx.x & 31;
    if (node >= n) return;
    float4 xv = ((const float4*)(X + (size_t)node * D))[lane];
    float4 sv = ((const float4*)vs)[lane];
    float4 dv = ((const float4*)vd)[lane];
    float ss = xv.x * sv.x + xv.y * sv.y + xv.z * sv.z + xv.w * sv.w;
    float sd = xv.x * dv.x + xv.y * dv.y + xv.z * dv.z + xv.w * dv.w;
    #pragma unroll
    for (int o = 16; o > 0; o >>= 1) {
        ss += __shfl_xor(ss, o, 32);
        sd += __shfl_xor(sd, o, 32);
    }
    if (lane == 0) { as_[node] = ss; ad_[node] = sd; }
}

__global__ void fill_kernel(float* __restrict__ p, float v, size_t n) {
    size_t i = blockIdx.x * (size_t)blockDim.x + threadIdx.x;
    if (i < n) p[i] = v;
}

__device__ inline void atomicMaxFloat(float* addr, float v) {
    if (v >= 0.f) atomicMax((int*)addr, __float_as_int(v));
    else          atomicMin((unsigned int*)addr, __float_as_uint(v));
}

__global__ void edge_max_kernel(const int* __restrict__ src, const int* __restrict__ dst,
                                const float* __restrict__ as_, const float* __restrict__ ad_,
                                float* __restrict__ emax, int E) {
    int e = blockIdx.x * blockDim.x + threadIdx.x;
    if (e >= E) return;
    float s = as_[src[e]] + ad_[dst[e]];
    s = (s >= 0.f) ? s : NEG_SLOPE * s;
    atomicMaxFloat(&emax[dst[e]], s);
}

__global__ void edge_exp_kernel(const int* __restrict__ src, const int* __restrict__ dst,
                                const float* __restrict__ as_, const float* __restrict__ ad_,
                                const float* __restrict__ emax, float* __restrict__ p,
                                float* __restrict__ denom, int E) {
    int e = blockIdx.x * blockDim.x + threadIdx.x;
    if (e >= E) return;
    int d = dst[e];
    float s = as_[src[e]] + ad_[d];
    s = (s >= 0.f) ? s : NEG_SLOPE * s;
    float m = emax[d];
    if (!(m > -3.0e38f && m < 3.0e38f)) m = 0.f;   // jnp.where(isfinite, m, 0)
    float pe = __expf(s - m);
    p[e] = pe;
    atomicAdd(&denom[d], pe);
}

// one wave per edge; float4 per lane; atomic f32 adds resolve in L2
__global__ __launch_bounds__(256) void edge_agg_kernel(
        const int* __restrict__ src, const int* __restrict__ dst,
        const float* __restrict__ p, const float* __restrict__ denom,
        const float* __restrict__ H, float* __restrict__ out, int E) {
    int e = (int)((blockIdx.x * (size_t)blockDim.x + threadIdx.x) >> 5);
    int lane = threadIdx.x & 31;
    if (e >= E) return;
    int s = src[e], d = dst[e];
    float alpha = p[e] / (denom[d] + 1e-16f);
    float4 h = ((const float4*)(H + (size_t)s * D))[lane];
    float* o = out + (size_t)d * D + lane * 4;
    atomicAdd(o + 0, alpha * h.x);
    atomicAdd(o + 1, alpha * h.y);
    atomicAdd(o + 2, alpha * h.z);
    atomicAdd(o + 3, alpha * h.w);
}

__global__ void bias_relu_kernel(float* __restrict__ buf, const float* __restrict__ b, size_t n) {
    size_t i = blockIdx.x * (size_t)blockDim.x + threadIdx.x;
    if (i >= n) return;
    float y = buf[i] + b[i & (D - 1)];
    buf[i] = y > 0.f ? y : 0.f;
}

__global__ __launch_bounds__(256) void pool_sum_kernel(
        const float* __restrict__ H, const int* __restrict__ batch,
        float* __restrict__ sums, float* __restrict__ cnt, int n) {
    int node = (int)((blockIdx.x * (size_t)blockDim.x + threadIdx.x) >> 5);
    int lane = threadIdx.x & 31;
    if (node >= n) return;
    int g = batch[node];
    float4 h = ((const float4*)(H + (size_t)node * D))[lane];
    float* s = sums + (size_t)g * D + lane * 4;
    atomicAdd(s + 0, h.x); atomicAdd(s + 1, h.y);
    atomicAdd(s + 2, h.z); atomicAdd(s + 3, h.w);
    if (lane == 0) atomicAdd(&cnt[g], 1.0f);
}

__global__ void pool_div_kernel(const float* __restrict__ sums, const float* __restrict__ cnt,
                                float* __restrict__ pooled, int n) {
    int i = blockIdx.x * blockDim.x + threadIdx.x;
    if (i >= n) return;
    float c = cnt[i >> 7];
    c = c > 1.f ? c : 1.f;
    pooled[i] = sums[i] / c;
}

__global__ void add_bias_kernel(float* __restrict__ y, const float* __restrict__ b, int n) {
    int i = blockIdx.x * blockDim.x + threadIdx.x;
    if (i >= n) return;
    y[i] += b[i & (D - 1)];
}

// ---------------------------------------------------------------------------

static inline void* bump(char*& p, size_t bytes) {
    void* r = (void*)p;
    p += (bytes + 255) & ~(size_t)255;
    return r;
}

extern "C" void kernel_launch(void* const* d_in, const int* in_sizes, int n_in,
                              void* d_out, int out_size, void* d_ws, size_t ws_size,
                              hipStream_t stream) {
    const float* x        = (const float*)d_in[0];
    const int*   ei       = (const int*)  d_in[1];
    const int*   batch    = (const int*)  d_in[2];
    const float* W_src1   = (const float*)d_in[3];
    const float* W_dst1   = (const float*)d_in[4];
    const float* att_src1 = (const float*)d_in[5];
    const float* att_dst1 = (const float*)d_in[6];
    const float* bias1    = (const float*)d_in[7];
    const float* W_src2   = (const float*)d_in[8];
    const float* W_dst2   = (const float*)d_in[9];
    const float* att_src2 = (const float*)d_in[10];
    const float* att_dst2 = (const float*)d_in[11];
    const float* bias2    = (const float*)d_in[12];
    const float* W_lin    = (const float*)d_in[13];
    const float* b_lin    = (const float*)d_in[14];
    float* out = (float*)d_out;

    const int N = in_sizes[0] / D;      // 50000
    const int E = in_sizes[1] / 2;      // 800000
    const int G = 64;
    const int* src = ei;
    const int* dst = ei + E;

    // workspace carve-out
    char* wsp = (char*)d_ws;
    _Float16* Wp1 = (_Float16*)bump(wsp, (size_t)D * D * sizeof(_Float16));
    _Float16* Wp2 = (_Float16*)bump(wsp, (size_t)D * D * sizeof(_Float16));
    _Float16* WpL = (_Float16*)bump(wsp, (size_t)D * D * sizeof(_Float16));
    float* vs1   = (float*)bump(wsp, D * sizeof(float));
    float* vd1   = (float*)bump(wsp, D * sizeof(float));
    float* vs2   = (float*)bump(wsp, D * sizeof(float));
    float* vd2   = (float*)bump(wsp, D * sizeof(float));
    float* as_   = (float*)bump(wsp, (size_t)N * sizeof(float));
    float* ad_   = (float*)bump(wsp, (size_t)N * sizeof(float));
    float* emax  = (float*)bump(wsp, (size_t)N * sizeof(float));
    float* denom = (float*)bump(wsp, (size_t)N * sizeof(float));
    float* pbuf  = (float*)bump(wsp, (size_t)E * sizeof(float));
    float* bufA  = (float*)bump(wsp, (size_t)N * D * sizeof(float)); // h_src (GEMM out)
    float* bufB  = (float*)bump(wsp, (size_t)N * D * sizeof(float)); // aggregate / layer out
    float* sums  = (float*)bump(wsp, (size_t)G * D * sizeof(float));
    float* cnt   = (float*)bump(wsp, (size_t)G * sizeof(float));
    float* pooled= (float*)bump(wsp, (size_t)G * D * sizeof(float));

    const int rowTiles = N / 16;                 // 3125
    const int gemmBlk  = (rowTiles + 7) / 8;     // 8 waves/block
    const int nodeWaveBlk = (N * 32 + 255) / 256;
    const int edgeBlk     = (E + 255) / 256;
    const int edgeWaveBlk = ((size_t)E * 32 + 255) / 256;
    const int elemBlk     = ((size_t)N * D + 255) / 256;

    // --- weight prep (packed f16 B-operands + folded attention vectors) ---
    pack_w_kernel<<<(D * D + 255) / 256, 256, 0, stream>>>(W_src1, Wp1);
    pack_w_kernel<<<(D * D + 255) / 256, 256, 0, stream>>>(W_src2, Wp2);
    pack_w_kernel<<<(D * D + 255) / 256, 256, 0, stream>>>(W_lin,  WpL);
    watt_kernel<<<1, 128, 0, stream>>>(W_src1, att_src1, vs1);
    watt_kernel<<<1, 128, 0, stream>>>(W_dst1, att_dst1, vd1);
    watt_kernel<<<1, 128, 0, stream>>>(W_src2, att_src2, vs2);
    watt_kernel<<<1, 128, 0, stream>>>(W_dst2, att_dst2, vd2);

    // ---------------- layer 1 ----------------
    gemm_wmma_kernel<<<gemmBlk, 256, 0, stream>>>(x, Wp1, bufA, N);        // h_src1
    node_scores_kernel<<<nodeWaveBlk, 256, 0, stream>>>(x, vs1, vd1, as_, ad_, N);
    fill_kernel<<<(N + 255) / 256, 256, 0, stream>>>(emax, -INFINITY, (size_t)N);
    hipMemsetAsync(denom, 0, (size_t)N * sizeof(float), stream);
    hipMemsetAsync(bufB, 0, (size_t)N * D * sizeof(float), stream);
    edge_max_kernel<<<edgeBlk, 256, 0, stream>>>(src, dst, as_, ad_, emax, E);
    edge_exp_kernel<<<edgeBlk, 256, 0, stream>>>(src, dst, as_, ad_, emax, pbuf, denom, E);
    edge_agg_kernel<<<edgeWaveBlk, 256, 0, stream>>>(src, dst, pbuf, denom, bufA, bufB, E);
    bias_relu_kernel<<<elemBlk, 256, 0, stream>>>(bufB, bias1, (size_t)N * D);

    // ---------------- layer 2 ----------------
    gemm_wmma_kernel<<<gemmBlk, 256, 0, stream>>>(bufB, Wp2, bufA, N);     // h_src2
    node_scores_kernel<<<nodeWaveBlk, 256, 0, stream>>>(bufB, vs2, vd2, as_, ad_, N);
    fill_kernel<<<(N + 255) / 256, 256, 0, stream>>>(emax, -INFINITY, (size_t)N);
    hipMemsetAsync(denom, 0, (size_t)N * sizeof(float), stream);
    hipMemsetAsync(bufB, 0, (size_t)N * D * sizeof(float), stream);        // h1 dead now
    edge_max_kernel<<<edgeBlk, 256, 0, stream>>>(src, dst, as_, ad_, emax, E);
    edge_exp_kernel<<<edgeBlk, 256, 0, stream>>>(src, dst, as_, ad_, emax, pbuf, denom, E);
    edge_agg_kernel<<<edgeWaveBlk, 256, 0, stream>>>(src, dst, pbuf, denom, bufA, bufB, E);
    bias_relu_kernel<<<elemBlk, 256, 0, stream>>>(bufB, bias2, (size_t)N * D);

    // ---------------- mean pool + head ----------------
    hipMemsetAsync(sums, 0, (size_t)G * D * sizeof(float), stream);
    hipMemsetAsync(cnt,  0, (size_t)G * sizeof(float), stream);
    pool_sum_kernel<<<nodeWaveBlk, 256, 0, stream>>>(bufB, batch, sums, cnt, N);
    pool_div_kernel<<<(G * D + 255) / 256, 256, 0, stream>>>(sums, cnt, pooled, G * D);
    gemm_wmma_kernel<<<1, 256, 0, stream>>>(pooled, WpL, out, G);          // 4 row tiles
    add_bias_kernel<<<(G * D + 255) / 256, 256, 0, stream>>>(out, b_lin, G * D);
}
